// VectorQuantizer_41412074668463
// MI455X (gfx1250) — matrix operand order repaired and checked by reading, not compiled, not run on hardware
//
#include <hip/hip_runtime.h>
#include <stdint.h>

// ---------------------------------------------------------------------------
// VectorQuantizer forward for MI455X (gfx1250, wave32).
// out[t, g, :] = codebook[g, argmin_k ||x[t,g,:] - codebook[g,k,:]||^2, :]
// Cross terms via V_WMMA_F32_16X16X32_BF16 with split-bf16 (hi/lo) inputs:
//   x*c ~= xh*ch + xh*cl + xl*ch   (fp32-class accuracy, bf16 WMMA rate)
// Codebook is staged global -> LDS with GLOBAL_LOAD_ASYNC_TO_LDS_B128
// (ASYNCcnt, double-buffered, overlapped with WMMA compute); all 8 waves of
// a block read the shared stage via ds_load_b128.
// Workspace layout (needs ~2.04 MB):
//   [0,      1 MB) : codebook bf16 hi plane  [8][1024][64]
//   [1 MB,   2 MB) : codebook bf16 lo plane  [8][1024][64]
//   [2 MB, +32 KB) : c_norm fp32             [8][1024]
// ---------------------------------------------------------------------------

typedef __attribute__((ext_vector_type(16))) __bf16    v16bf;
typedef __attribute__((ext_vector_type(8)))  float     v8f;

struct Frag {
    union {
        uint32_t u[8];
        v16bf    v;
    };
};

__device__ __forceinline__ uint32_t rne_bf16_bits(float f) {
    uint32_t u = __float_as_uint(f);
    return (u + 0x7FFFu + ((u >> 16) & 1u)) >> 16;
}

// Split f into bf16 hi + bf16 lo (residual), RNE both times.
__device__ __forceinline__ void split_bf16(float f, uint32_t& hi, uint32_t& lo) {
    uint32_t h = rne_bf16_bits(f);
    float fh = __uint_as_float(h << 16);
    uint32_t l = rne_bf16_bits(f - fh);
    hi = h;
    lo = l;
}

// Pack two floats into (hi-plane u32, lo-plane u32); a -> [15:0], b -> [31:16].
__device__ __forceinline__ void split2(float a, float b, uint32_t& ph, uint32_t& pl) {
    uint32_t ha, la, hb, lb;
    split_bf16(a, ha, la);
    split_bf16(b, hb, lb);
    ph = ha | (hb << 16);
    pl = la | (lb << 16);
}

__device__ __forceinline__ void fillfrag(Frag& f, uint4 a, uint4 b) {
    f.u[0] = a.x; f.u[1] = a.y; f.u[2] = a.z; f.u[3] = a.w;
    f.u[4] = b.x; f.u[5] = b.y; f.u[6] = b.z; f.u[7] = b.w;
}

__device__ __forceinline__ v8f wmma_bf16(v16bf a, v16bf b, v8f c) {
    // (neg_a, A, neg_b, B, c_mod, C, reuse_a, reuse_b)
    return __builtin_amdgcn_wmma_f32_16x16x32_bf16(
        false, a, false, b, (short)0, c, false, false);
}

// Async global -> LDS copy of 16 bytes (per lane); tracked by ASYNCcnt.
__device__ __forceinline__ void async_copy_b128(uint32_t lds_off, uint64_t gaddr) {
    asm volatile("global_load_async_to_lds_b128 %0, %1, off"
                 :: "v"(lds_off), "v"(gaddr)
                 : "memory");
}

// ---------------------------------------------------------------------------
// Prep: codebook fp32 [8][1024][64] -> bf16 hi/lo planes + fp32 row norms.
// One wave per code row; 8 waves (256 threads) per block; 1024 blocks.
// ---------------------------------------------------------------------------
__global__ __launch_bounds__(256) void vq_cb_prep(
    const float* __restrict__ cb,
    uint16_t* __restrict__ cbhi,
    uint16_t* __restrict__ cblo,
    float* __restrict__ cnorm)
{
    const int lane = threadIdx.x & 31;
    const int row  = blockIdx.x * 8 + (threadIdx.x >> 5);   // 0..8191 = g*1024+k

    const float* src = cb + (size_t)row * 64;
    float a = src[2 * lane + 0];
    float b = src[2 * lane + 1];

    uint32_t ph, pl;
    split2(a, b, ph, pl);
    ((uint32_t*)cbhi)[(size_t)row * 32 + lane] = ph;
    ((uint32_t*)cblo)[(size_t)row * 32 + lane] = pl;

    float s = a * a + b * b;
#pragma unroll
    for (int off = 16; off >= 1; off >>= 1)
        s += __shfl_xor(s, off, 32);
    if (lane == 0)
        cnorm[row] = s;
}

// ---------------------------------------------------------------------------
// Main: one wave handles 16 tokens x 1 group. 8 waves/block (same group,
// shared LDS-staged codebook). Grid = 1024 blocks:
//   g    = blockIdx.x >> 7           (8 groups)
//   tile = (blockIdx.x & 127)*8+wave (1024 tiles of 16 tokens)
// Codebook streamed in 8 stages of 128 codes; each stage = 16 KB hi + 16 KB
// lo in LDS; double buffered (64 KB total).
// ---------------------------------------------------------------------------

// Issue one stage's async copies: 256 threads move 32 KB (threads 0-127: hi
// plane, 128-255: lo plane; 128 B = 8 x b128 per thread).
__device__ __forceinline__ void stage_issue(
    const uint16_t* __restrict__ cbhi,
    const uint16_t* __restrict__ cblo,
    int g, int stage, uint32_t lds_buf_base, int tid)
{
    const int p  = tid >> 7;      // plane select
    const int tt = tid & 127;
    const uint16_t* src = (p ? cblo : cbhi)
                        + (size_t)(g * 1024 + stage * 128) * 64;
    const uint64_t ga = (uint64_t)(uintptr_t)((const char*)src + tt * 128);
    const uint32_t la = lds_buf_base + (uint32_t)p * 16384u + (uint32_t)tt * 128u;
#pragma unroll
    for (int j = 0; j < 8; ++j)
        async_copy_b128(la + j * 16, ga + (uint64_t)(j * 16));
}

__global__ __launch_bounds__(256) void vq_argmin(
    const float* __restrict__ x,          // [16384][8][64] fp32
    const float* __restrict__ cbf32,      // [8][1024][64] fp32
    const uint16_t* __restrict__ cbhi,    // [8][1024][64] bf16 hi
    const uint16_t* __restrict__ cblo,    // [8][1024][64] bf16 lo
    const float* __restrict__ cnorm,      // [8][1024] fp32
    float* __restrict__ out)              // [16384][8][64] fp32
{
    // [buf][plane][128 codes * 64 dims] halfs; 64 KB total.
    alignas(16) __shared__ uint16_t sh[2][2][128 * 64];

    const int tid    = threadIdx.x;
    const int lane   = tid & 31;
    const int wave   = tid >> 5;
    const int g      = blockIdx.x >> 7;
    const int tile   = (blockIdx.x & 127) * 8 + wave;
    const int token0 = tile * 16;
    const int h      = lane >> 4;     // lane half: K/row-offset selector
    const int c      = lane & 15;     // A: row index; B/C/D: column index

    const uint32_t sh_base = (uint32_t)(uintptr_t)&sh[0][0][0];

    // Kick off stage 0 copy before doing any other work.
    stage_issue(cbhi, cblo, g, 0, sh_base, tid);

    // ---- Build A fragments (x rows, split bf16), reused for all 64 chunks.
    // A 16x32 bf16, k order used here (as B): lane (h,c) holds row M=c; VGPR j: K=16h+2j(+1).
    // Two K-steps (d 0..31, 32..63).
    Frag Ahi[2], Alo[2];
    const float* xrow = x + (size_t)(token0 + c) * 512 + (size_t)g * 64;
#pragma unroll
    for (int s = 0; s < 2; ++s) {
#pragma unroll
        for (int part = 0; part < 2; ++part) {
            const float4* p = (const float4*)(xrow + s * 32 + 16 * h + part * 8);
            float4 f0 = p[0];
            float4 f1 = p[1];
            split2(f0.x, f0.y, Ahi[s].u[part * 4 + 0], Alo[s].u[part * 4 + 0]);
            split2(f0.z, f0.w, Ahi[s].u[part * 4 + 1], Alo[s].u[part * 4 + 1]);
            split2(f1.x, f1.y, Ahi[s].u[part * 4 + 2], Alo[s].u[part * 4 + 2]);
            split2(f1.z, f1.w, Ahi[s].u[part * 4 + 3], Alo[s].u[part * 4 + 3]);
        }
    }

    // ---- Running argmin state: VGPR v of C/D holds row M = v + 8h.
    float bestv[8];
    int   besti[8];
#pragma unroll
    for (int v = 0; v < 8; ++v) { bestv[v] = 3.4e38f; besti[v] = 0; }

    const size_t gbase = (size_t)g * 1024 * 64;
    const float* cn = cnorm + (size_t)g * 1024;

    // ---- 8 stages x 8 chunks of 16 codes.
    for (int s = 0; s < 8; ++s) {
        const int buf = s & 1;
        if (s + 1 < 8) {
            // Prefetch next stage into the other buffer, then wait for only
            // the previous 8 asyncs (ours complete in order).
            stage_issue(cbhi, cblo, g, s + 1, sh_base + (uint32_t)(buf ^ 1) * 32768u, tid);
            asm volatile("s_wait_asynccnt 0x8" ::: "memory");
        } else {
            asm volatile("s_wait_asynccnt 0x0" ::: "memory");
        }
        __syncthreads();   // all waves' stage-s data visible in LDS

        const uint16_t* shh = &sh[buf][0][0];
        const uint16_t* shl = &sh[buf][1][0];

        for (int nc = 0; nc < 8; ++nc) {
            const int code  = s * 128 + nc * 16 + c;
            const int codeL = nc * 16 + c;
            // B 32x16 bf16, k order used here (as A): lane (h,c) holds column c, K = 16h..16h+15,
            // i.e. d = kstep*32 + 16h .. +15 contiguous -> ds_load_b128 x2.
            const uint4* ph = (const uint4*)(shh + codeL * 64 + 16 * h);
            const uint4* pl = (const uint4*)(shl + codeL * 64 + 16 * h);
            Frag Bh0, Bh1, Bl0, Bl1;
            fillfrag(Bh0, ph[0], ph[1]);   // d  0..31 slice for this lane
            fillfrag(Bh1, ph[4], ph[5]);   // d 32..63 slice
            fillfrag(Bl0, pl[0], pl[1]);
            fillfrag(Bl1, pl[4], pl[5]);
            const float cnv = cn[code];

            v8f acc = (v8f)(0.0f);
            acc = wmma_bf16(Ahi[0].v, Bh0.v, acc);   // xh*ch
            acc = wmma_bf16(Ahi[1].v, Bh1.v, acc);
            acc = wmma_bf16(Ahi[0].v, Bl0.v, acc);   // xh*cl
            acc = wmma_bf16(Ahi[1].v, Bl1.v, acc);
            acc = wmma_bf16(Alo[0].v, Bh0.v, acc);   // xl*ch
            acc = wmma_bf16(Alo[1].v, Bh1.v, acc);

#pragma unroll
            for (int v = 0; v < 8; ++v) {
                float sc = fmaf(-2.0f, acc[v], cnv);  // ||c||^2 - 2 x.c
                if (sc < bestv[v]) { bestv[v] = sc; besti[v] = code; }
            }
        }
        __syncthreads();   // done reading buf before it gets overwritten
    }

    // ---- Min-reduce across the 16 lanes of each half (columns), with
    // first-index tie-break to match jnp.argmin semantics. Branchless.
#pragma unroll
    for (int v = 0; v < 8; ++v) {
        float bv = bestv[v];
        int   bi = besti[v];
#pragma unroll
        for (int off = 1; off < 16; off <<= 1) {
            float ov = __shfl_xor(bv, off, 32);
            int   oi = __shfl_xor(bi, off, 32);
            int take = (int)(ov < bv) | ((int)(ov == bv) & (int)(oi < bi));
            bv = take ? ov : bv;
            bi = take ? oi : bi;
        }
        besti[v] = bi;
    }

    // ---- Gather winning fp32 codebook rows. Half h owns tokens token0+8h+v;
    // 16 lanes x float4 cover the 64-float row.
    const float* cbg = cbf32 + gbase;
#pragma unroll
    for (int v = 0; v < 8; ++v) {
        const int token = token0 + 8 * h + v;
        float4 q = ((const float4*)(cbg + (size_t)besti[v] * 64))[c];
        ((float4*)(out + (size_t)token * 512 + (size_t)g * 64))[c] = q;
    }
}

// ---------------------------------------------------------------------------
extern "C" void kernel_launch(void* const* d_in, const int* in_sizes, int n_in,
                              void* d_out, int out_size, void* d_ws, size_t ws_size,
                              hipStream_t stream)
{
    (void)in_sizes; (void)n_in; (void)out_size; (void)ws_size;
    const float* x  = (const float*)d_in[0];   // [8,2048,512] fp32
    const float* cb = (const float*)d_in[1];   // [8,1024,64]  fp32

    uint16_t* cbhi = (uint16_t*)d_ws;                                   // 1 MB
    uint16_t* cblo = (uint16_t*)((char*)d_ws + (1u << 20));             // 1 MB
    float*    cnrm = (float*)((char*)d_ws + (2u << 20));                // 32 KB
    float*    out  = (float*)d_out;

    // 8192 codebook rows, 1 wave per row, 8 waves per block.
    vq_cb_prep<<<dim3(1024), dim3(256), 0, stream>>>(cb, cbhi, cblo, cnrm);

    // 8 groups x 1024 token-tiles; 8 waves per block share a group.
    vq_argmin<<<dim3(1024), dim3(256), 0, stream>>>(x, cb, cbhi, cblo, cnrm, out);
}
